// QuantizedMultiHeadAttention_85529978733202
// MI455X (gfx1250) — 16-bit and 8-bit matrix operand order repaired and checked by reading, compiled, not run on hardware
//
#include <hip/hip_runtime.h>
#include <hip/hip_bf16.h>
#include <math.h>


typedef __attribute__((ext_vector_type(16))) _Float16 v16h;
typedef __attribute__((ext_vector_type(8)))  _Float16 v8h;
typedef __attribute__((ext_vector_type(4)))  _Float16 v4h;
typedef __attribute__((ext_vector_type(8)))  float    v8f;
typedef __attribute__((ext_vector_type(8)))  int      v8i;
typedef __attribute__((ext_vector_type(4)))  int      v4i;
typedef __attribute__((ext_vector_type(2)))  int      v2i;
typedef unsigned char u8t;

#define NTHREADS 256
#define NB_RED   256
#define BK       32
#define BK8      64

struct QP { float mn, mx, scale, zp; };

__device__ __forceinline__ float fq_apply(float v, float s, float zp) {
    float q = rintf(v / s) + zp;
    q = fminf(fmaxf(q, 0.f), 255.f);
    return (q - zp) * s;
}
__device__ __forceinline__ unsigned int q_u8(float v, float s, float zp) {
    float q = rintf(v / s) + zp;
    return (unsigned int)fminf(fmaxf(q, 0.f), 255.f);
}

__device__ __forceinline__ void async_copy_b128(uint32_t lds_off, const void* gaddr) {
    asm volatile("global_load_async_to_lds_b128 %0, %1, off"
                 :: "v"(lds_off), "v"(gaddr)
                 : "memory");
}
__device__ __forceinline__ void wait_asynccnt0() {
    asm volatile("s_wait_asynccnt 0x0" ::: "memory");
}
__device__ __forceinline__ uint32_t lds_off32(const void* p) {
    return (uint32_t)(uintptr_t)p;
}

__global__ __launch_bounds__(NTHREADS) void reduce_minmax_kernel(
    const float* __restrict__ x, long long n, float* __restrict__ part) {
    __shared__ float smn[NTHREADS], smx[NTHREADS];
    int tid = threadIdx.x;
    float mn = 0.f, mx = 0.f;
    long long stride = (long long)gridDim.x * NTHREADS;
    for (long long i = (long long)blockIdx.x * NTHREADS + tid; i < n; i += stride) {
        float v = x[i];
        mn = fminf(mn, v);
        mx = fmaxf(mx, v);
    }
    smn[tid] = mn; smx[tid] = mx;
    __syncthreads();
    for (int s = NTHREADS / 2; s > 0; s >>= 1) {
        if (tid < s) {
            smn[tid] = fminf(smn[tid], smn[tid + s]);
            smx[tid] = fmaxf(smx[tid], smx[tid + s]);
        }
        __syncthreads();
    }
    if (tid == 0) {
        part[blockIdx.x]             = smn[0];
        part[gridDim.x + blockIdx.x] = smx[0];
    }
}

__global__ __launch_bounds__(NTHREADS) void finalize_params_kernel(
    const float* __restrict__ part, int nb, QP* __restrict__ params, int idx) {
    __shared__ float smn[NTHREADS], smx[NTHREADS];
    int tid = threadIdx.x;
    float mn = 0.f, mx = 0.f;
    for (int i = tid; i < nb; i += NTHREADS) {
        mn = fminf(mn, part[i]);
        mx = fmaxf(mx, part[nb + i]);
    }
    smn[tid] = mn; smx[tid] = mx;
    __syncthreads();
    for (int s = NTHREADS / 2; s > 0; s >>= 1) {
        if (tid < s) {
            smn[tid] = fminf(smn[tid], smn[tid + s]);
            smx[tid] = fmaxf(smx[tid], smx[tid + s]);
        }
        __syncthreads();
    }
    if (tid == 0) {
        mn = smn[0]; mx = smx[0];
        float scale = (mx - mn) / 255.f;
        if (!(scale > 0.f)) scale = 1.f;
        float zp = fminf(fmaxf(rintf(-mn / scale), 0.f), 255.f);
        params[idx] = {mn, mx, scale, zp};
    }
}

__global__ __launch_bounds__(NTHREADS) void chan_minmax_kernel(
    const float* __restrict__ x, int rows, QP* __restrict__ chanP) {
    __shared__ float smn[NTHREADS], smx[NTHREADS];
    int c = blockIdx.x;
    int tid = threadIdx.x;
    float mn = 0.f, mx = 0.f;
    int total = rows * 16;
    for (int i = tid; i < total; i += NTHREADS) {
        int r = i >> 4, h = i & 15;
        float v = x[(long long)r * 1024 + h * 64 + c];
        mn = fminf(mn, v);
        mx = fmaxf(mx, v);
    }
    smn[tid] = mn; smx[tid] = mx;
    __syncthreads();
    for (int s = NTHREADS / 2; s > 0; s >>= 1) {
        if (tid < s) {
            smn[tid] = fminf(smn[tid], smn[tid + s]);
            smx[tid] = fmaxf(smx[tid], smx[tid + s]);
        }
        __syncthreads();
    }
    if (tid == 0) {
        mn = smn[0]; mx = smx[0];
        float scale = (mx - mn) / 255.f;
        if (!(scale > 0.f)) scale = 1.f;
        float zp = fminf(fmaxf(rintf(-mn / scale), 0.f), 255.f);
        chanP[c] = {mn, mx, scale, zp};
    }
}

__global__ __launch_bounds__(NTHREADS) void fq8_pt_kernel(
    const float* __restrict__ x, u8t* __restrict__ y, long long n4,
    const QP* __restrict__ params, int idx) {
    QP p = params[idx];
    long long stride = (long long)gridDim.x * NTHREADS;
    for (long long i = (long long)blockIdx.x * NTHREADS + threadIdx.x; i < n4; i += stride) {
        float4 v = ((const float4*)x)[i];
        unsigned int pk = q_u8(v.x, p.scale, p.zp)
                        | (q_u8(v.y, p.scale, p.zp) << 8)
                        | (q_u8(v.z, p.scale, p.zp) << 16)
                        | (q_u8(v.w, p.scale, p.zp) << 24);
        ((unsigned int*)y)[i] = pk;
    }
}

__global__ __launch_bounds__(NTHREADS) void fq_pt_kernel(
    const float* __restrict__ x, _Float16* __restrict__ y, long long n4,
    const QP* __restrict__ params, int idx) {
    QP p = params[idx];
    long long stride = (long long)gridDim.x * NTHREADS;
    for (long long i = (long long)blockIdx.x * NTHREADS + threadIdx.x; i < n4; i += stride) {
        float4 v = ((const float4*)x)[i];
        v4h o;
        o[0] = (_Float16)fq_apply(v.x, p.scale, p.zp);
        o[1] = (_Float16)fq_apply(v.y, p.scale, p.zp);
        o[2] = (_Float16)fq_apply(v.z, p.scale, p.zp);
        o[3] = (_Float16)fq_apply(v.w, p.scale, p.zp);
        ((v4h*)y)[i] = o;
    }
}

__global__ __launch_bounds__(NTHREADS) void fq_chan_kernel(
    const float* __restrict__ x, _Float16* __restrict__ y, long long n4,
    const QP* __restrict__ chanP) {
    long long stride = (long long)gridDim.x * NTHREADS;
    for (long long i = (long long)blockIdx.x * NTHREADS + threadIdx.x; i < n4; i += stride) {
        float4 v = ((const float4*)x)[i];
        int c0 = (int)((i * 4) & 63);
        v4h o;
        QP p0 = chanP[c0 + 0]; o[0] = (_Float16)fq_apply(v.x, p0.scale, p0.zp);
        QP p1 = chanP[c0 + 1]; o[1] = (_Float16)fq_apply(v.y, p1.scale, p1.zp);
        QP p2 = chanP[c0 + 2]; o[2] = (_Float16)fq_apply(v.z, p2.scale, p2.zp);
        QP p3 = chanP[c0 + 3]; o[3] = (_Float16)fq_apply(v.w, p3.scale, p3.zp);
        ((v4h*)y)[i] = o;
    }
}

__global__ __launch_bounds__(NTHREADS) void rowsum_u8_kernel(
    const u8t* __restrict__ x, float* __restrict__ sums) {
    __shared__ int sred[NTHREADS];
    int tid = threadIdx.x;
    const u8t* p = x + (long long)blockIdx.x * 1024;
    unsigned int w = *(const unsigned int*)(p + tid * 4);
    int acc = (int)(w & 255) + (int)((w >> 8) & 255) +
              (int)((w >> 16) & 255) + (int)(w >> 24);
    sred[tid] = acc;
    __syncthreads();
    for (int s = NTHREADS / 2; s > 0; s >>= 1) {
        if (tid < s) sred[tid] += sred[tid + s];
        __syncthreads();
    }
    if (tid == 0) sums[blockIdx.x] = (float)sred[0];
}

__device__ __forceinline__ v16h load_fragA16(const _Float16* row, int hv) {
    v8h lo = *(const v8h*)(row + hv * 8);
    v8h hi = *(const v8h*)(row + 16 + hv * 8);
    return __builtin_shufflevector(lo, hi, 0, 1, 2, 3, 4, 5, 6, 7,
                                   8, 9, 10, 11, 12, 13, 14, 15);
}
__device__ __forceinline__ v16h load_fragB16(const _Float16* col, int hv) {
    v8h lo = *(const v8h*)(col + hv * 8);
    v8h hi = *(const v8h*)(col + 16 + hv * 8);
    return __builtin_shufflevector(lo, hi, 0, 1, 2, 3, 4, 5, 6, 7,
                                   8, 9, 10, 11, 12, 13, 14, 15);
}
__device__ __forceinline__ v8i load_fragA8(const u8t* row, int hv) {
    v4i lo = *(const v4i*)(row + hv * 16);
    v4i hi = *(const v4i*)(row + 32 + hv * 16);
    return __builtin_shufflevector(lo, hi, 0, 1, 2, 3, 4, 5, 6, 7);
}
__device__ __forceinline__ v8i load_fragB8(const u8t* col, int hv) {
    v4i lo = *(const v4i*)(col + hv * 16);
    v4i hi = *(const v4i*)(col + 32 + hv * 16);
    return __builtin_shufflevector(lo, hi, 0, 1, 2, 3, 4, 5, 6, 7);
}

#define GBM 64
#define GBN 128
__global__ __launch_bounds__(NTHREADS) void gemm_iu8_kernel(
    const u8t* __restrict__ A, const u8t* __restrict__ Bt,
    float* __restrict__ C, const float* __restrict__ bias,
    const float* __restrict__ sumA, const float* __restrict__ sumB,
    const QP* __restrict__ params, int ix, int iw,
    int K, int lda, int ldb, int ldc) {
    __shared__ __align__(16) u8t As[2][GBM][BK8 + 16];
    __shared__ __align__(16) u8t Bs[2][GBN][BK8 + 16];

    int tid = threadIdx.x;
    int wave = tid >> 5, lane = tid & 31;
    int hv = lane >> 4, l16 = lane & 15;

    int m0 = blockIdx.x * GBM;
    int n0 = blockIdx.y * GBN;
    int wm = (wave >> 2) * 32;
    int wn = (wave & 3) * 32;

    v8i acc[2][2] = {};

    int arow = tid >> 2, ag = tid & 3;
    int br1 = (tid + NTHREADS) >> 2;
    const u8t* aSrc  = A  + (long long)(m0 + arow) * lda + ag * 16;
    const u8t* bSrc0 = Bt + (long long)(n0 + arow) * ldb + ag * 16;
    const u8t* bSrc1 = Bt + (long long)(n0 + br1) * ldb + ag * 16;

    auto preload = [&](int buf, int kk) {
        async_copy_b128(lds_off32(&As[buf][arow][ag * 16]), aSrc + kk);
        async_copy_b128(lds_off32(&Bs[buf][arow][ag * 16]), bSrc0 + kk);
        async_copy_b128(lds_off32(&Bs[buf][br1][ag * 16]),  bSrc1 + kk);
    };

    preload(0, 0);
    int buf = 0;
    for (int kk = 0; kk < K; kk += BK8) {
        wait_asynccnt0();
        __syncthreads();
        if (kk + BK8 < K) preload(buf ^ 1, kk + BK8);

        v8i af[2], bf[2];
#pragma unroll
        for (int t = 0; t < 2; t++) {
            af[t] = load_fragA8(&As[buf][wm + t * 16 + l16][0], hv);
            bf[t] = load_fragB8(&Bs[buf][wn + t * 16 + l16][0], hv);
        }
#pragma unroll
        for (int tm = 0; tm < 2; tm++)
#pragma unroll
            for (int tn = 0; tn < 2; tn++)
                acc[tm][tn] = __builtin_amdgcn_wmma_i32_16x16x64_iu8(
                    false, af[tm], false, bf[tn], acc[tm][tn], false, false);
        buf ^= 1;
    }

    QP px = params[ix], pw = params[iw];
    float ssc = px.scale * pw.scale;
    float kzz = (float)K * px.zp * pw.zp;
#pragma unroll
    for (int tm = 0; tm < 2; tm++)
#pragma unroll
        for (int tn = 0; tn < 2; tn++) {
            int n = n0 + wn + tn * 16 + l16;
            float sw = sumB[n], bv = bias[n];
#pragma unroll
            for (int r = 0; r < 8; r++) {
                int m = m0 + wm + tm * 16 + r + 8 * hv;
                float dot = (float)acc[tm][tn][r]
                          - pw.zp * sumA[m] - px.zp * sw + kzz;
                C[(long long)m * ldc + n] = dot * ssc + bv;
            }
        }
}

__global__ __launch_bounds__(NTHREADS) void gemm_f16_kernel(
    const _Float16* __restrict__ A, const _Float16* __restrict__ Bt,
    float* __restrict__ C,
    int K, int lda, int ldb, int ldc,
    long long aOuter, long long aInner, long long bOuter, long long bInner,
    long long cStride, int innerCount, float alpha) {
    __shared__ __align__(16) _Float16 As[2][GBM][BK + 8];
    __shared__ __align__(16) _Float16 Bs[2][GBN][BK + 8];

    int tid = threadIdx.x;
    int wave = tid >> 5, lane = tid & 31;
    int hv = lane >> 4, l16 = lane & 15;

    int z = blockIdx.z;
    int zo = z / innerCount, zi = z - zo * innerCount;
    const _Float16* Az = A + zo * aOuter + zi * aInner;
    const _Float16* Bz = Bt + zo * bOuter + zi * bInner;
    float* Cz = C + (long long)z * cStride;

    int m0 = blockIdx.x * GBM;
    int n0 = blockIdx.y * GBN;
    int wm = (wave >> 2) * 32;
    int wn = (wave & 3) * 32;

    v8f acc[2][2] = {};

    int arow = tid >> 2, ag = tid & 3;
    int br1 = (tid + NTHREADS) >> 2;
    const _Float16* aSrc  = Az + (long long)(m0 + arow) * lda + ag * 8;
    const _Float16* bSrc0 = Bz + (long long)(n0 + arow) * ldb + ag * 8;
    const _Float16* bSrc1 = Bz + (long long)(n0 + br1) * ldb + ag * 8;

    auto preload = [&](int buf, int kk) {
        async_copy_b128(lds_off32(&As[buf][arow][ag * 8]), aSrc + kk);
        async_copy_b128(lds_off32(&Bs[buf][arow][ag * 8]), bSrc0 + kk);
        async_copy_b128(lds_off32(&Bs[buf][br1][ag * 8]),  bSrc1 + kk);
    };

    preload(0, 0);
    int buf = 0;
    for (int kk = 0; kk < K; kk += BK) {
        wait_asynccnt0();
        __syncthreads();
        if (kk + BK < K) preload(buf ^ 1, kk + BK);

        v16h af[2], bf[2];
#pragma unroll
        for (int t = 0; t < 2; t++) {
            af[t] = load_fragA16(&As[buf][wm + t * 16 + l16][0], hv);
            bf[t] = load_fragB16(&Bs[buf][wn + t * 16 + l16][0], hv);
        }
#pragma unroll
        for (int tm = 0; tm < 2; tm++)
#pragma unroll
            for (int tn = 0; tn < 2; tn++)
                acc[tm][tn] = __builtin_amdgcn_wmma_f32_16x16x32_f16(
                    false, af[tm], false, bf[tn], (short)0, acc[tm][tn], false, false);
        buf ^= 1;
    }

#pragma unroll
    for (int tm = 0; tm < 2; tm++)
#pragma unroll
        for (int tn = 0; tn < 2; tn++) {
            int n = n0 + wn + tn * 16 + l16;
#pragma unroll
            for (int r = 0; r < 8; r++) {
                int m = m0 + wm + tm * 16 + r + 8 * hv;
                Cz[(long long)m * ldc + n] = acc[tm][tn][r] * alpha;
            }
        }
}

__global__ __launch_bounds__(NTHREADS) void softmax_kernel(float* __restrict__ a) {
    __shared__ float sred[NTHREADS];
    int tid = threadIdx.x;
    float* p = a + (long long)blockIdx.x * 1024;
    float4 v = ((const float4*)p)[tid];

    float m = fmaxf(fmaxf(v.x, v.y), fmaxf(v.z, v.w));
    sred[tid] = m; __syncthreads();
    for (int s = NTHREADS / 2; s > 0; s >>= 1) {
        if (tid < s) sred[tid] = fmaxf(sred[tid], sred[tid + s]);
        __syncthreads();
    }
    float rowmax = sred[0]; __syncthreads();

    v.x = __expf(v.x - rowmax);
    v.y = __expf(v.y - rowmax);
    v.z = __expf(v.z - rowmax);
    v.w = __expf(v.w - rowmax);

    sred[tid] = v.x + v.y + v.z + v.w; __syncthreads();
    for (int s = NTHREADS / 2; s > 0; s >>= 1) {
        if (tid < s) sred[tid] += sred[tid + s];
        __syncthreads();
    }
    float inv = 1.f / sred[0];
    v.x *= inv; v.y *= inv; v.z *= inv; v.w *= inv;
    ((float4*)p)[tid] = v;
}

#define ABM 128
#define ABN 64
__global__ __launch_bounds__(NTHREADS) void attn_av_kernel(
    const float* __restrict__ Aprob, const _Float16* __restrict__ V16,
    float* __restrict__ O, const QP* __restrict__ params, int pidx) {
    __shared__ __align__(16) _Float16 As[ABM][BK + 8];
    __shared__ __align__(16) _Float16 Bs[ABN][BK + 8];

    QP p = params[pidx];
    float s = p.scale, zp = p.zp;

    int tid = threadIdx.x;
    int wave = tid >> 5, lane = tid & 31;
    int hv = lane >> 4, l16 = lane & 15;
    int z = blockIdx.z, b = z >> 4, h = z & 15;

    const float* Az = Aprob + (long long)z * (1024LL * 1024LL);
    const _Float16* Bz = V16 + (long long)b * (1024LL * 1024LL) + h * 64;
    float* Oz = O + (long long)b * (1024LL * 1024LL) + h * 64;

    int m0 = blockIdx.x * ABM;
    int wm = (wave >> 1) * 32;
    int wn = (wave & 1) * 32;

    v8f acc[2][2] = {};

    int arow = tid >> 1, ag = tid & 1;
    int brow = tid >> 3, bg = tid & 7;

    for (int kk = 0; kk < 1024; kk += BK) {
        const float* src = Az + (long long)(m0 + arow) * 1024 + kk + ag * 16;
        float4 va = ((const float4*)src)[0];
        float4 vb = ((const float4*)src)[1];
        float4 vc = ((const float4*)src)[2];
        float4 vd = ((const float4*)src)[3];
        v8h bv = *(const v8h*)(Bz + (long long)(kk + brow) * 1024 + bg * 8);

        _Float16* d = &As[arow][ag * 16];
        d[0]  = (_Float16)fq_apply(va.x, s, zp);
        d[1]  = (_Float16)fq_apply(va.y, s, zp);
        d[2]  = (_Float16)fq_apply(va.z, s, zp);
        d[3]  = (_Float16)fq_apply(va.w, s, zp);
        d[4]  = (_Float16)fq_apply(vb.x, s, zp);
        d[5]  = (_Float16)fq_apply(vb.y, s, zp);
        d[6]  = (_Float16)fq_apply(vb.z, s, zp);
        d[7]  = (_Float16)fq_apply(vb.w, s, zp);
        d[8]  = (_Float16)fq_apply(vc.x, s, zp);
        d[9]  = (_Float16)fq_apply(vc.y, s, zp);
        d[10] = (_Float16)fq_apply(vc.z, s, zp);
        d[11] = (_Float16)fq_apply(vc.w, s, zp);
        d[12] = (_Float16)fq_apply(vd.x, s, zp);
        d[13] = (_Float16)fq_apply(vd.y, s, zp);
        d[14] = (_Float16)fq_apply(vd.z, s, zp);
        d[15] = (_Float16)fq_apply(vd.w, s, zp);
#pragma unroll
        for (int j = 0; j < 8; j++) Bs[bg * 8 + j][brow] = bv[j];
        __syncthreads();

        v16h af[2], bf[2];
#pragma unroll
        for (int t = 0; t < 2; t++) {
            af[t] = load_fragA16(&As[wm + t * 16 + l16][0], hv);
            bf[t] = load_fragB16(&Bs[wn + t * 16 + l16][0], hv);
        }
#pragma unroll
        for (int tm = 0; tm < 2; tm++)
#pragma unroll
            for (int tn = 0; tn < 2; tn++)
                acc[tm][tn] = __builtin_amdgcn_wmma_f32_16x16x32_f16(
                    false, af[tm], false, bf[tn], (short)0, acc[tm][tn], false, false);
        __syncthreads();
    }

#pragma unroll
    for (int tm = 0; tm < 2; tm++)
#pragma unroll
        for (int tn = 0; tn < 2; tn++) {
            int n = wn + tn * 16 + l16;
#pragma unroll
            for (int r = 0; r < 8; r++) {
                int m = m0 + wm + tm * 16 + r + 8 * hv;
                Oz[(long long)m * 1024 + n] = acc[tm][tn][r];
            }
        }
}

extern "C" void kernel_launch(void* const* d_in, const int* in_sizes, int n_in,
                              void* d_out, int out_size, void* d_ws, size_t ws_size,
                              hipStream_t stream) {
    (void)in_sizes; (void)n_in; (void)out_size;
    const long long NX = 4LL * 1024 * 1024;
    const long long NW = 1024LL * 1024;
    const long long NA = 64LL * 1024 * 1024;

    const float* q    = (const float*)d_in[0];
    const float* k    = (const float*)d_in[1];
    const float* v    = (const float*)d_in[2];
    const float* Wq_w = (const float*)d_in[3];
    const float* Wq_b = (const float*)d_in[4];
    const float* Wk_w = (const float*)d_in[5];
    const float* Wk_b = (const float*)d_in[6];
    const float* Wv_w = (const float*)d_in[7];
    const float* Wv_b = (const float*)d_in[8];
    const float* Wo_w = (const float*)d_in[9];
    const float* Wo_b = (const float*)d_in[10];

    float* out_o = (float*)d_out;
    float* out_a = out_o + NX;

    const size_t MB = 1ull << 20;
    const size_t KB = 1ull << 10;
    if (ws_size < 82 * MB) return;

    char* ws = (char*)d_ws;
    QP*    params = (QP*)(ws);
    QP*    chanQ  = (QP*)(ws + 4096);
    QP*    chanK  = (QP*)(ws + 8192);
    float* part   = (float*)(ws + 12288);
    u8t* q8  = (u8t*)(ws + 1 * MB);
    u8t* k8  = (u8t*)(ws + 5 * MB);
    u8t* v8  = (u8t*)(ws + 9 * MB);
    u8t* Wq8 = (u8t*)(ws + 13 * MB);
    u8t* Wk8 = (u8t*)(ws + 14 * MB);
    u8t* Wv8 = (u8t*)(ws + 15 * MB);
    u8t* Wo8 = (u8t*)(ws + 16 * MB);
    float* sumXq = (float*)(ws + 17 * MB);
    float* sumXk = (float*)(ws + 17 * MB + 64 * KB);
    float* sumXv = (float*)(ws + 17 * MB + 128 * KB);
    float* sumXo = (float*)(ws + 17 * MB + 192 * KB);
    float* sumWq = (float*)(ws + 17 * MB + 256 * KB);
    float* sumWk = (float*)(ws + 17 * MB + 320 * KB);
    float* sumWv = (float*)(ws + 17 * MB + 384 * KB);
    float* sumWo = (float*)(ws + 17 * MB + 448 * KB);
    _Float16* qq16 = (_Float16*)(ws + 18 * MB);
    _Float16* kq16 = (_Float16*)(ws + 26 * MB);
    float* qlin = (float*)(ws + 34 * MB);
    float* klin = (float*)(ws + 50 * MB);
    float* vlin = (float*)(ws + 66 * MB);
    float*    o_mid = qlin;
    _Float16* vq16  = (_Float16*)klin;
    u8t*      o8    = q8;

    auto minmax = [&](const float* x, long long n, int idx) {
        reduce_minmax_kernel<<<NB_RED, NTHREADS, 0, stream>>>(x, n, part);
        finalize_params_kernel<<<1, NTHREADS, 0, stream>>>(part, NB_RED, params, idx);
    };

    minmax(q, NX, 0); minmax(k, NX, 1); minmax(v, NX, 2);
    minmax(Wq_w, NW, 3); minmax(Wk_w, NW, 4); minmax(Wv_w, NW, 5); minmax(Wo_w, NW, 6);

    fq8_pt_kernel<<<2048, NTHREADS, 0, stream>>>(q, q8, NX / 4, params, 0);
    fq8_pt_kernel<<<2048, NTHREADS, 0, stream>>>(k, k8, NX / 4, params, 1);
    fq8_pt_kernel<<<2048, NTHREADS, 0, stream>>>(v, v8, NX / 4, params, 2);
    fq8_pt_kernel<<<1024, NTHREADS, 0, stream>>>(Wq_w, Wq8, NW / 4, params, 3);
    fq8_pt_kernel<<<1024, NTHREADS, 0, stream>>>(Wk_w, Wk8, NW / 4, params, 4);
    fq8_pt_kernel<<<1024, NTHREADS, 0, stream>>>(Wv_w, Wv8, NW / 4, params, 5);
    fq8_pt_kernel<<<1024, NTHREADS, 0, stream>>>(Wo_w, Wo8, NW / 4, params, 6);
    rowsum_u8_kernel<<<4096, NTHREADS, 0, stream>>>(q8, sumXq);
    rowsum_u8_kernel<<<4096, NTHREADS, 0, stream>>>(k8, sumXk);
    rowsum_u8_kernel<<<4096, NTHREADS, 0, stream>>>(v8, sumXv);
    rowsum_u8_kernel<<<1024, NTHREADS, 0, stream>>>(Wq8, sumWq);
    rowsum_u8_kernel<<<1024, NTHREADS, 0, stream>>>(Wk8, sumWk);
    rowsum_u8_kernel<<<1024, NTHREADS, 0, stream>>>(Wv8, sumWv);
    rowsum_u8_kernel<<<1024, NTHREADS, 0, stream>>>(Wo8, sumWo);

    dim3 gLin(4096 / GBM, 1024 / GBN, 1);
    gemm_iu8_kernel<<<gLin, NTHREADS, 0, stream>>>(q8, Wq8, qlin, Wq_b,
        sumXq, sumWq, params, 0, 3, 1024, 1024, 1024, 1024);
    gemm_iu8_kernel<<<gLin, NTHREADS, 0, stream>>>(k8, Wk8, klin, Wk_b,
        sumXk, sumWk, params, 1, 4, 1024, 1024, 1024, 1024);
    gemm_iu8_kernel<<<gLin, NTHREADS, 0, stream>>>(v8, Wv8, vlin, Wv_b,
        sumXv, sumWv, params, 2, 5, 1024, 1024, 1024, 1024);

    chan_minmax_kernel<<<64, NTHREADS, 0, stream>>>(qlin, 4096, chanQ);
    chan_minmax_kernel<<<64, NTHREADS, 0, stream>>>(klin, 4096, chanK);
    minmax(vlin, NX, 7);
    fq_chan_kernel<<<2048, NTHREADS, 0, stream>>>(qlin, qq16, NX / 4, chanQ);
    fq_chan_kernel<<<2048, NTHREADS, 0, stream>>>(klin, kq16, NX / 4, chanK);
    fq_pt_kernel<<<2048, NTHREADS, 0, stream>>>(vlin, vq16, NX / 4, params, 7);

    dim3 gSc(1024 / GBM, 1024 / GBN, 64);
    gemm_f16_kernel<<<gSc, NTHREADS, 0, stream>>>(qq16, kq16, out_a,
        64, 1024, 1024, 1024,
        1024LL * 1024, 64, 1024LL * 1024, 64,
        1024LL * 1024, 16, 1.0f / 32.0f);

    softmax_kernel<<<65536, NTHREADS, 0, stream>>>(out_a);
    minmax(out_a, NA, 9);

    dim3 gAv(1024 / ABM, 1, 64);
    attn_av_kernel<<<gAv, NTHREADS, 0, stream>>>(out_a, vq16, o_mid, params, 9);

    minmax(o_mid, NX, 8);
    fq8_pt_kernel<<<2048, NTHREADS, 0, stream>>>(o_mid, o8, NX / 4, params, 8);
    rowsum_u8_kernel<<<4096, NTHREADS, 0, stream>>>(o8, sumXo);
    gemm_iu8_kernel<<<gLin, NTHREADS, 0, stream>>>(o8, Wo8, out_o, Wo_b,
        sumXo, sumWo, params, 8, 6, 1024, 1024, 1024, 1024);
}
